// TraceUnitHandshake_14963666059481
// MI455X (gfx1250) — matrix operand order repaired and checked by reading, not compiled, not run on hardware
//
#include <hip/hip_runtime.h>
#include <hip/hip_bf16.h>
#include <math.h>

// ---------------------------------------------------------------------------
// TraceUnitHandshake fused kernel for gfx1250 (MI455X), wave32 + WMMA bf16.
//  * Pre-pass: convert all weights f32 -> bf16, stored TRANSPOSED (W^T[n][k])
//    in d_ws so every WMMA B-fragment is one contiguous 32B vector load.
//  * Main kernel: one workgroup per (b,t); 8 waves; all GEMMs are
//    v_wmma_f32_16x16x32_bf16 with f32 accumulation.
// ---------------------------------------------------------------------------

typedef __attribute__((ext_vector_type(16))) __bf16 v16bf;
typedef __attribute__((ext_vector_type(8)))  __bf16 v8bf;
typedef __attribute__((ext_vector_type(8)))  float  v8f;

#define UDIM 384   // UNIT_DIM
#define HS   128   // head dim
#define MM    64   // dyn units
#define NN    64   // sem units

// ---- workspace layout (bf16 elements), all weights transposed W^T[n][k] ----
#define WS_WQT 0                       // [128][384]
#define WS_WKT (WS_WQT + 128 * 384)    // [128][384]
#define WS_WVT (WS_WKT + 128 * 384)    // [128][384]
#define WS_WOT (WS_WVT + 128 * 384)    // [384][128]
#define WS_WGT (WS_WOT + 384 * 128)    // [384][768]
#define WS_ELEMS (WS_WGT + 384 * 768)  // 491520 bf16 = 983040 bytes

// ---- dynamic LDS layout (bytes). Regions are time-aliased. ----
#define SZD_OFF    0                   // z_dyn bf16 64x384 (all phases)
#define SZS_OFF    49152               // z_sem bf16 64x384 (P0-P1); residual alias (P4-P5)
#define REGC_OFF   98304
#define SQ_OFF     (REGC_OFF + 0)      // Q bf16 64x128   (P1->P2)
#define SK_OFF     (REGC_OFF + 16384)  // K bf16 64x128   (P1->P2)
#define SVT_OFF    (REGC_OFF + 32768)  // V^T bf16 128x64 (P1->P3)
#define SLG_OFF    (REGC_OFF + 49152)  // logits f32 64x64 (P2)
#define SATTN_OFF  (REGC_OFF + 65536)  // attn bf16 64x64 (P2->P3)
#define SCTX_OFF   (REGC_OFF + 73728)  // ctx bf16 64x128 (P3->P4)
#define SX_OFF     (REGC_OFF + 0)      // x f32 64x384   (P5->P6, aliases regC)
#define SMEM_BYTES (REGC_OFF + 98304)  // 196608 B = 192 KB per workgroup

__device__ __forceinline__ v8f wmma_bf16(v16bf a, v16bf b, v8f c) {
  return __builtin_amdgcn_wmma_f32_16x16x32_bf16(false, a, false, b, (short)0, c,
                                                 false, false);
}

// A fragment: 16x32 bf16, row-major source (LDS), leading dim ld.
// lane L (r = L&15, kb = (L>>4)*16): f[j]=A[r][kb+j], f[8+j]=A[r][kb+8+j].
// Two contiguous 16B vector reads -> 2x ds_load_b128.
__device__ __forceinline__ v16bf load_A_lds(const __bf16* base, int ld) {
  const int lane = threadIdx.x & 31;
  const __bf16* p = base + (lane & 15) * ld + ((lane >> 4) * 16);
  v8bf lo = *(const v8bf*)(p);
  v8bf hi = *(const v8bf*)(p + 8);
  return __builtin_shufflevector(lo, hi, 0, 1, 2, 3, 4, 5, 6, 7,
                                 8, 9, 10, 11, 12, 13, 14, 15);
}

// B fragment where the logical B[k][c] = S[c][k] (transposed source),
// i.e. S is stored [column][k] row-major with leading dim ld.
// lane L (c = L&15, kb = (L>>4)*16): one contiguous 32B vector read.
__device__ __forceinline__ v16bf load_Bt_lds(const __bf16* base, int ld) {
  const int lane = threadIdx.x & 31;
  return *(const v16bf*)(base + (lane & 15) * ld + ((lane >> 4) * 16));
}

// B fragment from a GLOBAL bf16 transposed weight W^T[n][k] (leading dim ldk).
// f[j] = W[k0+kb+j][n0+c] = W^T[n0+c][k0+kb+j] : one contiguous 32B load.
__device__ __forceinline__ v16bf load_B_gmem_bf16t(const __bf16* baseT, int ldk) {
  const int lane = threadIdx.x & 31;
  return *(const v16bf*)(baseT + (lane & 15) * ldk + ((lane >> 4) * 16));
}

// Store 16x16 f32 C/D fragment as bf16, row-major (leading dim ld).
// D layout: VGPR v -> row v + 8*(lane>>4), col = lane&15.
__device__ __forceinline__ void store_C_bf16(__bf16* base, int ld, v8f acc) {
  const int lane = threadIdx.x & 31;
  const int n  = lane & 15;
  const int mh = (lane >> 4) * 8;
#pragma unroll
  for (int v = 0; v < 8; ++v) base[(v + mh) * ld + n] = (__bf16)acc[v];
}

// Store 16x16 f32 C/D fragment TRANSPOSED as bf16: dstT[col][row], leading dim
// ldT. Each lane's 8 rows are contiguous in dstT -> one packed 16B store.
__device__ __forceinline__ void store_C_bf16_T(__bf16* baseT, int ldT, v8f acc) {
  const int lane = threadIdx.x & 31;
  const int c  = lane & 15;
  const int mh = (lane >> 4) * 8;
  v8bf o;
#pragma unroll
  for (int v = 0; v < 8; ++v) o[v] = (__bf16)acc[v];
  *(v8bf*)(baseT + c * ldT + mh) = o;
}

// ---------------- Pre-pass: W[K][N] f32 -> W^T[N][K] bf16 -------------------
__global__ void transpose_to_bf16(const float* __restrict__ src,
                                  __bf16* __restrict__ dst, int K, int N) {
  const int i = blockIdx.x * blockDim.x + threadIdx.x;
  if (i < K * N) {
    const int n = i / K;
    const int k = i - n * K;
    dst[n * K + k] = (__bf16)src[k * N + n];
  }
}

// ---------------- Main fused kernel ----------------------------------------
__global__ void __launch_bounds__(256, 1)
handshake_kernel(const float* __restrict__ z_dyn, const float* __restrict__ z_sem,
                 const unsigned char* __restrict__ presence,
                 const __bf16* __restrict__ WqT, const float* __restrict__ bq,
                 const __bf16* __restrict__ WkT, const float* __restrict__ bk,
                 const __bf16* __restrict__ WvT, const float* __restrict__ bv,
                 const __bf16* __restrict__ WoT, const float* __restrict__ bo,
                 const __bf16* __restrict__ WgT, const float* __restrict__ bg,
                 const float* __restrict__ gamma, const float* __restrict__ beta,
                 float* __restrict__ out) {
  extern __shared__ __align__(16) char smem[];
  __bf16* sZD   = (__bf16*)(smem + SZD_OFF);
  __bf16* sZS   = (__bf16*)(smem + SZS_OFF);
  __bf16* sRes  = sZS;                       // alias (phase >= 4)
  __bf16* sQ    = (__bf16*)(smem + SQ_OFF);
  __bf16* sK    = (__bf16*)(smem + SK_OFF);
  __bf16* sVT   = (__bf16*)(smem + SVT_OFF); // V transposed: [128][64]
  float*  sLg   = (float*) (smem + SLG_OFF);
  __bf16* sAttn = (__bf16*)(smem + SATTN_OFF);
  __bf16* sCtx  = (__bf16*)(smem + SCTX_OFF);
  float*  sX    = (float*) (smem + SX_OFF);  // alias (phase >= 5)

  const int bt   = blockIdx.x;               // 0 .. B*T-1
  const int tid  = threadIdx.x;
  const int lane = tid & 31;
  const int wave = tid >> 5;                 // 0..7
  const int hi   = lane >> 4;

  const float* gzd = z_dyn + (size_t)bt * MM * UDIM;
  const float* gzs = z_sem + (size_t)bt * NN * UDIM;
  const unsigned char* up = presence + (size_t)bt * NN;

  // ---------------- Phase 0: stage activations as bf16 in LDS --------------
  {
    const float4* d4 = (const float4*)gzd;
    const float4* s4 = (const float4*)gzs;
    for (int i = tid; i < (MM * UDIM) / 4; i += 256) {
      float4 a = d4[i];
      sZD[4 * i + 0] = (__bf16)a.x; sZD[4 * i + 1] = (__bf16)a.y;
      sZD[4 * i + 2] = (__bf16)a.z; sZD[4 * i + 3] = (__bf16)a.w;
      float4 b = s4[i];
      sZS[4 * i + 0] = (__bf16)b.x; sZS[4 * i + 1] = (__bf16)b.y;
      sZS[4 * i + 2] = (__bf16)b.z; sZS[4 * i + 3] = (__bf16)b.w;
    }
  }
  __syncthreads();

  // ---------------- Phase 1: Q = zd*Wq+bq, K = zs*Wk+bk, V = zs*Wv+bv ------
  {
    const int r0 = (wave >> 1) * 16;
    const int c0 = (wave & 1) * 64;
    v8f aq[4], ak[4], av[4];
#pragma unroll
    for (int t = 0; t < 4; ++t) {
      v8f z = {0.f, 0.f, 0.f, 0.f, 0.f, 0.f, 0.f, 0.f};
      aq[t] = z; ak[t] = z; av[t] = z;
    }
    for (int k0 = 0; k0 < UDIM; k0 += 32) {
      v16bf aD = load_A_lds(sZD + r0 * UDIM + k0, UDIM);
      v16bf aS = load_A_lds(sZS + r0 * UDIM + k0, UDIM);
#pragma unroll
      for (int t = 0; t < 4; ++t) {
        const int n0 = c0 + 16 * t;
        aq[t] = wmma_bf16(aD, load_B_gmem_bf16t(WqT + n0 * UDIM + k0, UDIM), aq[t]);
        ak[t] = wmma_bf16(aS, load_B_gmem_bf16t(WkT + n0 * UDIM + k0, UDIM), ak[t]);
        av[t] = wmma_bf16(aS, load_B_gmem_bf16t(WvT + n0 * UDIM + k0, UDIM), av[t]);
      }
    }
#pragma unroll
    for (int t = 0; t < 4; ++t) {
      const int n = c0 + 16 * t + (lane & 15);
      const float vbq = bq[n], vbk = bk[n], vbv = bv[n];
#pragma unroll
      for (int v = 0; v < 8; ++v) { aq[t][v] += vbq; ak[t][v] += vbk; av[t][v] += vbv; }
      store_C_bf16(sQ + r0 * HS + c0 + 16 * t, HS, aq[t]);
      store_C_bf16(sK + r0 * HS + c0 + 16 * t, HS, ak[t]);
      // V stored transposed: sVT[d][n] so the context GEMM B-frag is contiguous.
      store_C_bf16_T(sVT + (c0 + 16 * t) * NN + r0, NN, av[t]);
    }
  }
  __syncthreads();

  // ---------------- Phase 2: logits = Q K^T * scale, mask ------------------
  {
    const int r0 = (wave >> 1) * 16;
    const int c0 = (wave & 1) * 32;             // 2 col tiles per wave
    const float scale = 0.08838834764831845f;   // 1/sqrt(128)
    v8f acc[2];
    v8f z = {0.f, 0.f, 0.f, 0.f, 0.f, 0.f, 0.f, 0.f};
    acc[0] = z; acc[1] = z;
    for (int d0 = 0; d0 < HS; d0 += 32) {
      v16bf aQ = load_A_lds(sQ + r0 * HS + d0, HS);
#pragma unroll
      for (int t = 0; t < 2; ++t)   // B[d][n] = K[n][d]: contiguous rows of sK
        acc[t] = wmma_bf16(aQ, load_Bt_lds(sK + (c0 + 16 * t) * HS + d0, HS), acc[t]);
    }
#pragma unroll
    for (int t = 0; t < 2; ++t) {
      const int n = c0 + 16 * t + (lane & 15);
      const bool p = up[n] != 0;
#pragma unroll
      for (int v = 0; v < 8; ++v) {
        const int row = r0 + v + 8 * hi;
        sLg[row * NN + n] = p ? acc[t][v] * scale : -1.0e9f;
      }
    }
  }
  __syncthreads();

  // ---------------- Phase 2b: row softmax + presence zeroing ---------------
  if (tid < MM) {
    const int row = tid;
    float mx = -INFINITY;
    for (int j = 0; j < NN; ++j) mx = fmaxf(mx, sLg[row * NN + j]);
    float s = 0.f;
    for (int j = 0; j < NN; ++j) s += __expf(sLg[row * NN + j] - mx);
    const float inv = 1.0f / s;
    for (int j = 0; j < NN; ++j) {
      const bool p = up[j] != 0;
      sAttn[row * NN + j] = p ? (__bf16)(__expf(sLg[row * NN + j] - mx) * inv)
                              : (__bf16)0.0f;
    }
  }
  __syncthreads();

  // ---------------- Phase 3: context = attn @ V ----------------------------
  {
    const int r0 = (wave >> 1) * 16;
    const int c0 = (wave & 1) * 64;
    v8f acc[4];
    v8f z = {0.f, 0.f, 0.f, 0.f, 0.f, 0.f, 0.f, 0.f};
#pragma unroll
    for (int t = 0; t < 4; ++t) acc[t] = z;
    for (int k0 = 0; k0 < NN; k0 += 32) {
      v16bf aA = load_A_lds(sAttn + r0 * NN + k0, NN);
#pragma unroll
      for (int t = 0; t < 4; ++t)   // B[n][d] = V[n][d] = sVT[d][n]: contiguous
        acc[t] = wmma_bf16(aA, load_Bt_lds(sVT + (c0 + 16 * t) * NN + k0, NN), acc[t]);
    }
#pragma unroll
    for (int t = 0; t < 4; ++t) store_C_bf16(sCtx + r0 * HS + c0 + 16 * t, HS, acc[t]);
  }
  __syncthreads();

  // ---------------- Phase 4: residual = context @ Wo + bo ------------------
  {
    const int r0 = (wave >> 1) * 16;
    const int c0 = (wave & 1) * 192;           // 12 col tiles per wave
    v8f acc[12];
    v8f z = {0.f, 0.f, 0.f, 0.f, 0.f, 0.f, 0.f, 0.f};
#pragma unroll
    for (int t = 0; t < 12; ++t) acc[t] = z;
    for (int k0 = 0; k0 < HS; k0 += 32) {
      v16bf aC = load_A_lds(sCtx + r0 * HS + k0, HS);
#pragma unroll
      for (int t = 0; t < 12; ++t)
        acc[t] = wmma_bf16(aC, load_B_gmem_bf16t(WoT + (c0 + 16 * t) * HS + k0, HS),
                           acc[t]);
    }
#pragma unroll
    for (int t = 0; t < 12; ++t) {
      const int n = c0 + 16 * t + (lane & 15);
      const float vb = bo[n];
#pragma unroll
      for (int v = 0; v < 8; ++v) acc[t][v] += vb;
      store_C_bf16(sRes + r0 * UDIM + c0 + 16 * t, UDIM, acc[t]);   // overwrites sZS
    }
  }
  __syncthreads();

  // ---------------- Phase 5: gate GEMM (768->384), sigmoid, x --------------
  // gate_lin = z_dyn @ Wg[0:384] + residual @ Wg[384:768] + bg
  {
    const int r0 = (wave >> 1) * 16;
    const int c0 = (wave & 1) * 192;
    v8f acc[12];
    v8f z = {0.f, 0.f, 0.f, 0.f, 0.f, 0.f, 0.f, 0.f};
#pragma unroll
    for (int t = 0; t < 12; ++t) acc[t] = z;
    for (int k0 = 0; k0 < UDIM; k0 += 32) {
      v16bf aD = load_A_lds(sZD + r0 * UDIM + k0, UDIM);
#pragma unroll
      for (int t = 0; t < 12; ++t)
        acc[t] = wmma_bf16(aD,
                           load_B_gmem_bf16t(WgT + (c0 + 16 * t) * (2 * UDIM) + k0,
                                             2 * UDIM),
                           acc[t]);
    }
    for (int k0 = 0; k0 < UDIM; k0 += 32) {
      v16bf aR = load_A_lds(sRes + r0 * UDIM + k0, UDIM);
#pragma unroll
      for (int t = 0; t < 12; ++t)
        acc[t] = wmma_bf16(aR,
                           load_B_gmem_bf16t(
                               WgT + (c0 + 16 * t) * (2 * UDIM) + UDIM + k0,
                               2 * UDIM),
                           acc[t]);
    }
#pragma unroll
    for (int t = 0; t < 12; ++t) {
      const int n = c0 + 16 * t + (lane & 15);
      const float vb = bg[n];
#pragma unroll
      for (int v = 0; v < 8; ++v) {
        const int row = r0 + v + 8 * hi;
        const float g   = 1.0f / (1.0f + __expf(-(acc[t][v] + vb)));
        const float res = (float)sRes[row * UDIM + n];
        const float zd  = (float)sZD[row * UDIM + n];
        sX[row * UDIM + n] = zd + g * res;      // aliases region C (now dead)
      }
    }
  }
  __syncthreads();

  // ---------------- Phase 6: LayerNorm over last dim -----------------------
  if (tid < MM) {
    const int row = tid;
    float s = 0.f, ss = 0.f;
    for (int j = 0; j < UDIM; ++j) {
      const float xv = sX[row * UDIM + j];
      s += xv; ss += xv * xv;
    }
    const float mu  = s * (1.0f / UDIM);
    const float var = ss * (1.0f / UDIM) - mu * mu;
    const float inv = rsqrtf(var + 1e-5f);
    float* o = out + (size_t)bt * MM * UDIM + (size_t)row * UDIM;
    for (int j = 0; j < UDIM; ++j)
      o[j] = (sX[row * UDIM + j] - mu) * inv * gamma[j] + beta[j];
  }
}

extern "C" void kernel_launch(void* const* d_in, const int* in_sizes, int n_in,
                              void* d_out, int out_size, void* d_ws, size_t ws_size,
                              hipStream_t stream) {
  (void)in_sizes; (void)n_in; (void)ws_size; (void)out_size;
  const float* z_dyn = (const float*)d_in[0];
  const float* z_sem = (const float*)d_in[1];
  const unsigned char* presence = (const unsigned char*)d_in[2];  // bool mask
  const float* Wq = (const float*)d_in[3];
  const float* bq = (const float*)d_in[4];
  const float* Wk = (const float*)d_in[5];
  const float* bk = (const float*)d_in[6];
  const float* Wv = (const float*)d_in[7];
  const float* bv = (const float*)d_in[8];
  const float* Wo = (const float*)d_in[9];
  const float* bo = (const float*)d_in[10];
  const float* Wg = (const float*)d_in[11];
  const float* bg = (const float*)d_in[12];
  const float* gamma = (const float*)d_in[13];
  const float* beta  = (const float*)d_in[14];
  float* out = (float*)d_out;

  __bf16* ws = (__bf16*)d_ws;
  __bf16* WqT = ws + WS_WQT;
  __bf16* WkT = ws + WS_WKT;
  __bf16* WvT = ws + WS_WVT;
  __bf16* WoT = ws + WS_WOT;
  __bf16* WgT = ws + WS_WGT;

  // Pre-pass: weights -> bf16, transposed (runs every call; deterministic).
  transpose_to_bf16<<<(384 * 128 + 255) / 256, 256, 0, stream>>>(Wq, WqT, 384, 128);
  transpose_to_bf16<<<(384 * 128 + 255) / 256, 256, 0, stream>>>(Wk, WkT, 384, 128);
  transpose_to_bf16<<<(384 * 128 + 255) / 256, 256, 0, stream>>>(Wv, WvT, 384, 128);
  transpose_to_bf16<<<(128 * 384 + 255) / 256, 256, 0, stream>>>(Wo, WoT, 128, 384);
  transpose_to_bf16<<<(768 * 384 + 255) / 256, 256, 0, stream>>>(Wg, WgT, 768, 384);

  // 192 KB dynamic LDS per workgroup (WGP has 320 KB) — raise the cap.
  (void)hipFuncSetAttribute((const void*)handshake_kernel,
                            hipFuncAttributeMaxDynamicSharedMemorySize, SMEM_BYTES);

  const int grid = 4 * 256;   // B * T
  handshake_kernel<<<grid, 256, SMEM_BYTES, stream>>>(
      z_dyn, z_sem, presence, WqT, bq, WkT, bk, WvT, bv, WoT, bo, WgT, bg,
      gamma, beta, out);
}
